// MultiViewSelfAttention_86887188398876
// MI455X (gfx1250) — matrix operand order repaired and checked by reading, not compiled, not run on hardware
//
#include <hip/hip_runtime.h>
#include <hip/hip_bf16.h>

#define VIEW 8
#define SEQ  2048
#define HID  1024
#define ATT  128

typedef __bf16 bf16_t;
typedef __attribute__((ext_vector_type(16))) __bf16 v16bf;
typedef __attribute__((ext_vector_type(8)))  __bf16 v8bf;
typedef __attribute__((ext_vector_type(8)))  float  v8f;
typedef __attribute__((ext_vector_type(4)))  float  v4f;
typedef __attribute__((ext_vector_type(4)))  unsigned int v4u;
typedef __attribute__((ext_vector_type(8)))  int    v8i;
typedef __attribute__((ext_vector_type(4)))  int    v4i;

static __device__ __forceinline__ v8f wmma_bf16(v16bf a, v16bf b, v8f c) {
  // (neg_a, A, neg_b, B, c_mod, C, reuse_a, reuse_b)
  return __builtin_amdgcn_wmma_f32_16x16x32_bf16(false, a, false, b, (short)0, c,
                                                 false, false);
}

// A-operand (16x32 MxK bf16): lane m=L&15, hi=L>>4 holds K in {8hi..8hi+7} U {16+8hi..16+8hi+7}
static __device__ __forceinline__ v16bf a_from_bf16_row(const bf16_t* row, int lane) {
  const int hi = (lane >> 4) & 1;
  v8bf lo = *(const v8bf*)(row + 8 * hi);
  v8bf up = *(const v8bf*)(row + 16 + 8 * hi);
  v16bf r;
#pragma unroll
  for (int i = 0; i < 8; ++i) { r[i] = lo[i]; r[8 + i] = up[i]; }
  return r;
}

static __device__ __forceinline__ v16bf a_from_f32_row(const float* row, int lane) {
  const int hi = (lane >> 4) & 1;
  const float* p0 = row + 8 * hi;
  const float* p1 = row + 16 + 8 * hi;
  v4f a0 = *(const v4f*)p0, a1 = *(const v4f*)(p0 + 4);
  v4f b0 = *(const v4f*)p1, b1 = *(const v4f*)(p1 + 4);
  v16bf r;
#pragma unroll
  for (int i = 0; i < 4; ++i) {
    r[i]      = (__bf16)a0[i];
    r[4 + i]  = (__bf16)a1[i];
    r[8 + i]  = (__bf16)b0[i];
    r[12 + i] = (__bf16)b1[i];
  }
  return r;
}

// B-operand (32x16 KxN bf16): lane n=L&15, kh=L>>4 holds K in {8kh..8kh+7} U {16+8kh..16+8kh+7}.
static __device__ __forceinline__ v16bf b_from_bf16(const bf16_t* colrow, int lane) {
  const int kh = (lane >> 4) & 1;
  v8bf lo = *(const v8bf*)(colrow + 8 * kh);
  v8bf up = *(const v8bf*)(colrow + 16 + 8 * kh);
  v16bf r;
#pragma unroll
  for (int i = 0; i < 8; ++i) { r[i] = lo[i]; r[8 + i] = up[i]; }
  return r;
}

// ---- TDM: async DMA of a 16x32 f32 tile (row stride SEQ) from global into LDS ----
// D# per CDNA5 ISA ch.8: group0 = {flags/count, lds_addr, global_addr lo/hi|type},
// group1 = {mask/data_size, dims, tile dims, strides}. 2D tensor -> trailing groups zero.
static __device__ __forceinline__ void tdm_load_bias_tile(const float* gptr,
                                                          unsigned lds_off) {
  unsigned long long ga = (unsigned long long)(uintptr_t)gptr;
  v4u g0;
  g0[0] = 1u;                                               // count=1 (valid), user mode
  g0[1] = lds_off;                                          // lds_addr (bytes)
  g0[2] = (unsigned)(ga & 0xFFFFFFFFu);                     // global_addr[31:0]
  g0[3] = (unsigned)((ga >> 32) & 0x1FFFFFFu) | (2u << 30); // global_addr[56:32] | type=2
  v8i g1;
  g1[0] = (int)(2u << 16);                                  // data_size = 4B
  g1[1] = (int)((SEQ & 0xFFFF) << 16);                      // tensor_dim0[15:0]
  g1[2] = (int)(((SEQ >> 16) & 0xFFFF) | ((SEQ & 0xFFFF) << 16)); // dim0 hi | dim1 lo
  g1[3] = (int)(((SEQ >> 16) & 0xFFFF) | (32u << 16));      // dim1 hi | tile_dim0=32
  g1[4] = 16;                                               // tile_dim1=16, tile_dim2=0
  g1[5] = SEQ;                                              // tensor_dim0_stride lo32
  g1[6] = 0;
  g1[7] = 0;
  v4i z4 = {};
  v8i z8 = {};
  asm volatile("" ::: "memory");
  __builtin_amdgcn_tensor_load_to_lds(g0, g1, z4, z4, z8, 0);
}

// ---------------- weight convert + transpose: dst[c*R + r] = bf16(src[r*C + c]) -------------
__global__ __launch_bounds__(256) void transpose_cvt_kernel(const float* __restrict__ src,
                                                            bf16_t* __restrict__ dst,
                                                            int R, int C) {
  int idx = blockIdx.x * 256 + threadIdx.x;
  if (idx >= R * C) return;
  int r = idx / C, c = idx - r * C;
  dst[(size_t)c * R + r] = (__bf16)src[(size_t)r * C + c];
}

// ---------------- fused QKV projection: qh/kh [V][S][A] bf16, vh transposed [V][A][S] -------
__global__ __launch_bounds__(256) void proj_kernel(
    const float* __restrict__ q, const float* __restrict__ k, const float* __restrict__ v,
    const bf16_t* __restrict__ wqT, const bf16_t* __restrict__ wkT, const bf16_t* __restrict__ wvT,
    const float* __restrict__ bq, const float* __restrict__ bk, const float* __restrict__ bv,
    bf16_t* __restrict__ qh, bf16_t* __restrict__ kh, bf16_t* __restrict__ vhT) {
  const int lane = threadIdx.x & 31;
  const int wid  = threadIdx.x >> 5;   // 8 waves -> 8 ATT tiles
  const int s0   = blockIdx.x * 16;
  const int a0   = wid * 16;
  const int mat  = blockIdx.z % 3;
  const int view = blockIdx.z / 3;

  const float* X; const bf16_t* WT; const float* bias;
  if (mat == 0)      { X = q; WT = wqT; bias = bq; }
  else if (mat == 1) { X = k; WT = wkT; bias = bk; }
  else               { X = v; WT = wvT; bias = bv; }

  const float*  xrow = X  + ((size_t)view * SEQ + s0 + (lane & 15)) * HID;
  const bf16_t* wrow = WT + (size_t)(a0 + (lane & 15)) * HID;

  v8f acc = {};
#pragma unroll 4
  for (int k0 = 0; k0 < HID; k0 += 32)
    acc = wmma_bf16(a_from_f32_row(xrow + k0, lane), b_from_bf16(wrow + k0, lane), acc);

  const int n = lane & 15, mh = lane >> 4;
  const float bn = bias[a0 + n];
  if (mat == 2) {
    v8bf pk;                                // rows s0+8mh .. s0+8mh+7 contiguous in vhT
#pragma unroll
    for (int t = 0; t < 8; ++t) pk[t] = (__bf16)(acc[t] + bn);
    *(v8bf*)(vhT + ((size_t)view * ATT + a0 + n) * SEQ + s0 + 8 * mh) = pk;
  } else {
    bf16_t* dst     = (mat == 0) ? qh : kh;
    const float scl = (mat == 0) ? 0.08838834764831845f : 1.0f;  // ATT^-0.5 folded into qh
#pragma unroll
    for (int t = 0; t < 8; ++t)
      dst[((size_t)view * SEQ + s0 + 8 * mh + t) * ATT + a0 + n] = (__bf16)((acc[t] + bn) * scl);
  }
}

// ---------------- flash attention with post-softmax bias: x = softmax(qh khT) vh + bias vh ---
__global__ __launch_bounds__(32) void attn_kernel(
    const bf16_t* __restrict__ qh, const bf16_t* __restrict__ kh,
    const bf16_t* __restrict__ vhT, const float* __restrict__ attn_bias,
    bf16_t* __restrict__ xout) {
  __shared__ __align__(16) bf16_t pbuf[16 * 32];
  __shared__ __align__(16) float  btile[2][16 * 32];   // TDM double buffer (bias tiles)

  const int lane = threadIdx.x;
  const int q0   = blockIdx.x * 16;
  const int view = blockIdx.y;
  const int nlo  = lane & 15, mh = lane >> 4;

  const bf16_t* qbase = qh  + (size_t)view * SEQ * ATT;
  const bf16_t* kbase = kh  + (size_t)view * SEQ * ATT;
  const bf16_t* vtb   = vhT + (size_t)view * ATT * SEQ;
  const float*  bbase = attn_bias + ((size_t)view * SEQ + q0) * SEQ;

  const unsigned blds0 = (unsigned)(uintptr_t)&btile[0][0];  // LDS offset = addr[31:0]
  const unsigned blds1 = (unsigned)(uintptr_t)&btile[1][0];

  // Prime the TDM pipeline with the first bias tile
  tdm_load_bias_tile(bbase, blds0);

  // Preload the 16x128 qh tile as four A operands (stays in VGPRs for all 64 key tiles)
  v16bf qa[4];
  {
    const bf16_t* qrow = qbase + (size_t)(q0 + nlo) * ATT;
#pragma unroll
    for (int c = 0; c < 4; ++c) qa[c] = a_from_bf16_row(qrow + 32 * c, lane);
  }

  v8f U[8] = {}, W[8] = {};
  float mrun[8], lrun[8];
#pragma unroll
  for (int t = 0; t < 8; ++t) { mrun[t] = -__builtin_inff(); lrun[t] = 0.f; }

  for (int k0 = 0; k0 < SEQ; k0 += 32) {
    const int it = k0 >> 5;
    const bool more = (k0 + 32) < SEQ;
    // Issue the async DMA for the *next* bias tile before doing this tile's math
    if (more) tdm_load_bias_tile(bbase + (k0 + 32), (it & 1) ? blds0 : blds1);

    // ---- score tile 16q x 32k (two N halves), K = ATT = 128 ----
    v8f sc0 = {}, sc1 = {};
#pragma unroll
    for (int c = 0; c < 4; ++c) {
      v16bf b0 = b_from_bf16(kbase + (size_t)(k0 + nlo) * ATT + 32 * c, lane);
      v16bf b1 = b_from_bf16(kbase + (size_t)(k0 + 16 + nlo) * ATT + 32 * c, lane);
      sc0 = wmma_bf16(qa[c], b0, sc0);
      sc1 = wmma_bf16(qa[c], b1, sc1);
    }
    // ---- online softmax stats over the 32 new keys ----
    float alpha[8];
#pragma unroll
    for (int t = 0; t < 8; ++t) {
      float mx = fmaxf(sc0[t], sc1[t]);
#pragma unroll
      for (int off = 1; off < 16; off <<= 1) mx = fmaxf(mx, __shfl_xor(mx, off, 32));
      float mnew = fmaxf(mrun[t], mx);
      alpha[t]   = __expf(mrun[t] - mnew);
      sc0[t]     = __expf(sc0[t] - mnew);
      sc1[t]     = __expf(sc1[t] - mnew);
      float sm = sc0[t] + sc1[t];
#pragma unroll
      for (int off = 1; off < 16; off <<= 1) sm += __shfl_xor(sm, off, 32);
      lrun[t] = lrun[t] * alpha[t] + sm;
      mrun[t] = mnew;
    }
    // rescale running U
#pragma unroll
    for (int t = 0; t < 8; ++t) {
#pragma unroll
      for (int vv = 0; vv < 8; ++vv) U[t][vv] *= alpha[vv];
    }
    // ---- C-layout P -> A-layout via per-wave LDS staging ----
    __syncthreads();
#pragma unroll
    for (int t = 0; t < 8; ++t) {
      pbuf[(t + 8 * mh) * 32 + nlo]      = (__bf16)sc0[t];
      pbuf[(t + 8 * mh) * 32 + 16 + nlo] = (__bf16)sc1[t];
    }
    __syncthreads();
    v16bf pa = a_from_bf16_row(&pbuf[nlo * 32], lane);

    // ---- bias tile: wait for this iteration's TDM DMA, read from LDS ----
    if (more) {
      __builtin_amdgcn_s_wait_tensorcnt(1);   // current tile done, next still in flight
    } else {
      __builtin_amdgcn_s_wait_tensorcnt(0);   // drain on the last tile
    }
    asm volatile("" ::: "memory");
    v16bf ba = a_from_f32_row(&btile[it & 1][nlo * 32], lane);

    // ---- accumulate U += P*vh, W += bias*vh over the 8 ATT tiles (shared B) ----
#pragma unroll
    for (int t = 0; t < 8; ++t) {
      v16bf vb = b_from_bf16(vtb + (size_t)(16 * t + nlo) * SEQ + k0, lane);
      U[t] = wmma_bf16(pa, vb, U[t]);
      W[t] = wmma_bf16(ba, vb, W[t]);
    }
  }

  // ---- epilogue: x[s][view*ATT + a] = U/l + W, already in concat-transposed layout ----
#pragma unroll
  for (int t = 0; t < 8; ++t) {
#pragma unroll
    for (int vv = 0; vv < 8; ++vv) {
      float val = U[t][vv] / lrun[vv] + W[t][vv];
      xout[(size_t)(q0 + vv + 8 * mh) * (VIEW * ATT) + view * ATT + 16 * t + nlo] = (__bf16)val;
    }
  }
}

// ---------------- output projection: out = x @ wo + bo (f32 result) -----------------------
__global__ __launch_bounds__(256) void outproj_kernel(const bf16_t* __restrict__ x,
                                                      const bf16_t* __restrict__ woT,
                                                      const float* __restrict__ bo,
                                                      float* __restrict__ out) {
  const int lane = threadIdx.x & 31;
  const int wid  = threadIdx.x >> 5;
  const int s0   = blockIdx.x * 16;
  const int n0   = (blockIdx.y * 8 + wid) * 16;

  const bf16_t* xrow = x   + (size_t)(s0 + (lane & 15)) * HID;
  const bf16_t* wrow = woT + (size_t)(n0 + (lane & 15)) * HID;

  v8f acc = {};
#pragma unroll 4
  for (int k0 = 0; k0 < HID; k0 += 32)
    acc = wmma_bf16(a_from_bf16_row(xrow + k0, lane), b_from_bf16(wrow + k0, lane), acc);

  const int n = lane & 15, mh = lane >> 4;
  const float bn = bo[n0 + n];
#pragma unroll
  for (int t = 0; t < 8; ++t)
    out[(size_t)(s0 + t + 8 * mh) * HID + n0 + n] = acc[t] + bn;
}

extern "C" void kernel_launch(void* const* d_in, const int* in_sizes, int n_in,
                              void* d_out, int out_size, void* d_ws, size_t ws_size,
                              hipStream_t stream) {
  const float* q         = (const float*)d_in[0];
  const float* k         = (const float*)d_in[1];
  const float* v         = (const float*)d_in[2];
  const float* attn_bias = (const float*)d_in[3];
  const float* wq        = (const float*)d_in[4];
  const float* bq        = (const float*)d_in[5];
  const float* wk        = (const float*)d_in[6];
  const float* bk        = (const float*)d_in[7];
  const float* wv        = (const float*)d_in[8];
  const float* bv        = (const float*)d_in[9];
  const float* wo        = (const float*)d_in[10];
  const float* bo        = (const float*)d_in[11];
  float* out             = (float*)d_out;

  // bf16 workspace layout (~19.7 MB)
  bf16_t* ws  = (bf16_t*)d_ws;
  bf16_t* wqT = ws;                               // [ATT][HID]
  bf16_t* wkT = wqT + (size_t)ATT * HID;
  bf16_t* wvT = wkT + (size_t)ATT * HID;
  bf16_t* woT = wvT + (size_t)ATT * HID;          // [HID][HID]
  bf16_t* qh  = woT + (size_t)HID * HID;          // [V][S][ATT]
  bf16_t* kh  = qh  + (size_t)VIEW * SEQ * ATT;
  bf16_t* vhT = kh  + (size_t)VIEW * SEQ * ATT;   // [V][ATT][S]
  bf16_t* xb  = vhT + (size_t)VIEW * SEQ * ATT;   // [S][V*ATT]

  transpose_cvt_kernel<<<(HID * ATT + 255) / 256, 256, 0, stream>>>(wq, wqT, HID, ATT);
  transpose_cvt_kernel<<<(HID * ATT + 255) / 256, 256, 0, stream>>>(wk, wkT, HID, ATT);
  transpose_cvt_kernel<<<(HID * ATT + 255) / 256, 256, 0, stream>>>(wv, wvT, HID, ATT);
  transpose_cvt_kernel<<<(HID * HID + 255) / 256, 256, 0, stream>>>(wo, woT, HID, HID);

  proj_kernel<<<dim3(SEQ / 16, 1, VIEW * 3), 256, 0, stream>>>(
      q, k, v, wqT, wkT, wvT, bq, bk, bv, qh, kh, vhT);

  attn_kernel<<<dim3(SEQ / 16, VIEW), 32, 0, stream>>>(qh, kh, vhT, attn_bias, xb);

  outproj_kernel<<<dim3(SEQ / 16, HID / 16 / 8), 256, 0, stream>>>(xb, woT, bo, out);
}
